// TopKWindowAttentionLayer_27839978012830
// MI455X (gfx1250) — matrix operand order repaired and checked by reading, not compiled, not run on hardware
//
#include <hip/hip_runtime.h>
#include <hip/hip_bf16.h>

// ---------------------------------------------------------------------------
// TopK window attention block for MI455X (gfx1250), wave32 + WMMA bf16.
// NHWC activation layouts: every WMMA B-fragment is one contiguous 32B load
// per lane (contiguous k = 16*half + s order, same in A), A-fragments are two
// 16B loads. Each GEMM wave computes a 64x32 C tile (4 M-tiles x 2 N-tiles):
// A fragments shared across N, B fragments shared across M -> 1.5 VMEM per
// WMMA and two independent WMMA streams for load/compute overlap.
// ---------------------------------------------------------------------------

#define BB   8
#define DD   256
#define HH   56
#define WW   56
#define HWSZ 3136          // 56*56
#define HEADS 8
#define DH   32
#define WIN  7
#define NWND 64
#define WWIN 49
#define TOPK 8
#define KLEN 392           // TOPK*WWIN
#define KPAD 416           // 13*32
#define DM   1024
#define EPS  1e-5f

typedef __attribute__((ext_vector_type(8)))  unsigned short v8u;
typedef __attribute__((ext_vector_type(16))) unsigned short v16u;
typedef __attribute__((ext_vector_type(16))) __bf16        v16bf;
typedef __attribute__((ext_vector_type(8)))  float         v8f;

// ------------------------- scalar helpers ---------------------------------
__device__ __forceinline__ unsigned short f2bf(float f) {
  union { float f; unsigned int u; } c; c.f = f;
  unsigned int u = c.u;
  return (unsigned short)((u + 0x7FFFu + ((u >> 16) & 1u)) >> 16);   // RNE
}
__device__ __forceinline__ float bf2f(unsigned short h) {
  union { unsigned int u; float f; } c; c.u = ((unsigned int)h) << 16;
  return c.f;
}
__device__ __forceinline__ float gelu_exact(float x) {
  return 0.5f * x * (1.0f + erff(x * 0.70710678118654752f));
}
__device__ __forceinline__ float bnv(float x, float g, float b, float m, float v) {
  return g * (x - m) * rsqrtf(v + EPS) + b;
}

// ------------------------- WMMA helpers -----------------------------------
__device__ __forceinline__ v8f wmma_bf16(v16u a, v16u b, v8f c) {
  return __builtin_amdgcn_wmma_f32_16x16x32_bf16(
      false, __builtin_bit_cast(v16bf, a),
      false, __builtin_bit_cast(v16bf, b),
      (short)0, c, false, false);
}
__device__ __forceinline__ v16u concat8(v8u lo, v8u hi) {
  v16u r;
#pragma unroll
  for (int j = 0; j < 8; ++j) { r[j] = lo[j]; r[8 + j] = hi[j]; }
  return r;
}

// 64x32 C tile GEMM core (4 M-tiles x 2 N-tiles per wave).
// W: row-major [64 x KDIM] (row offset applied).  Bcol0/Bcol1: column
// pointers for n = n0+lm and n0+16+lm, K contiguous (NHWC).
// acc[mt*2+ntile]: C rows mt*16 + half*8 + i, col (ntile*16) + lm.
template <int KDIM>
__device__ __forceinline__ void gemm_tile4x2(const unsigned short* __restrict__ W,
                                             const unsigned short* __restrict__ Bcol0,
                                             const unsigned short* __restrict__ Bcol1,
                                             v8f acc[8], int lane) {
  const int half = (lane >> 4) & 1, lm = lane & 15;
  for (int kk = 0; kk < KDIM; kk += 32) {
    v16u bf0 = *(const v16u*)(Bcol0 + kk + half * 16);   // contiguous-k B fragments (k = 16*half + s)
    v16u bf1 = *(const v16u*)(Bcol1 + kk + half * 16);
#pragma unroll
    for (int mt = 0; mt < 4; ++mt) {
      const unsigned short* Ap = W + (long)(mt * 16 + lm) * KDIM + kk;
      v16u af = concat8(*(const v8u*)(Ap + half * 16),
                        *(const v8u*)(Ap + half * 16 + 8));
      acc[mt * 2 + 0] = wmma_bf16(af, bf0, acc[mt * 2 + 0]);
      acc[mt * 2 + 1] = wmma_bf16(af, bf1, acc[mt * 2 + 1]);
    }
  }
}

// ------------------------- elementwise kernels ----------------------------
__global__ void k_f32_to_bf16(const float* __restrict__ src,
                              unsigned short* __restrict__ dst, int n) {
  int i = blockIdx.x * 256 + threadIdx.x;
  if (i < n) dst[i] = f2bf(src[i]);
}

// NCHW x -> NHWC: x_t (raw f32, residual source) and h = bf16(gelu(bn1(x)))
__global__ void k_bn_gelu_x(const float* __restrict__ x,
                            const float* __restrict__ g, const float* __restrict__ b,
                            const float* __restrict__ m, const float* __restrict__ v,
                            float* __restrict__ x_t, unsigned short* __restrict__ h) {
  long i = (long)blockIdx.x * 256 + threadIdx.x;          // [b][hw][c], c fastest
  if (i >= (long)BB * HWSZ * DD) return;
  int c = (int)(i & (DD - 1));
  long t = i >> 8;                  // b*HWSZ + hw
  int hw = (int)(t % HWSZ);
  int bb = (int)(t / HWSZ);
  float val = x[((long)bb * DD + c) * HWSZ + hw];
  x_t[i] = val;
  h[i] = f2bf(gelu_exact(bnv(val, g[c], b[c], m[c], v[c])));
}

// ------------------------- qkv GEMM ---------------------------------------
// qkv[b][hw][768] = W_qkv(768x256) * h[b][hw][256]
__global__ void __launch_bounds__(32)
k_gemm_qkv(const unsigned short* __restrict__ Wq,
           const unsigned short* __restrict__ hB,
           unsigned short* __restrict__ qkv) {
  const int lane = threadIdx.x & 31;
  const int half = lane >> 4, lm = lane & 15;
  const int n0 = blockIdx.x * 32, m0 = blockIdx.y * 64, b = blockIdx.z;
  const unsigned short* Bcol0 = hB + ((long)b * HWSZ + n0 + lm) * DD;
  const unsigned short* Bcol1 = Bcol0 + (long)16 * DD;
  v8f acc[8] = {};
  gemm_tile4x2<DD>(Wq + (long)m0 * DD, Bcol0, Bcol1, acc, lane);
#pragma unroll
  for (int nt = 0; nt < 2; ++nt) {
    unsigned short* orow =
        qkv + ((long)b * HWSZ + n0 + nt * 16 + lm) * (3 * DD) + m0 + half * 8;
#pragma unroll
    for (int mt = 0; mt < 4; ++mt) {
      v8u o;
#pragma unroll
      for (int i = 0; i < 8; ++i) o[i] = f2bf(acc[mt * 2 + nt][i]);
      *(v8u*)(orow + mt * 16) = o;
    }
  }
}

// ------------------------- routing ----------------------------------------
__global__ void k_pool(const unsigned short* __restrict__ qkv,
                       float* __restrict__ qr, float* __restrict__ kr) {
  int bn = blockIdx.x;            // b*64 + n
  int b = bn >> 6, n = bn & 63;
  int c = threadIdx.x;            // 0..255
  int wy = n >> 3, wx = n & 7;
  float sq = 0.f, sk = 0.f;
  for (int r = 0; r < WWIN; ++r) {
    int hw = (wy * WIN + r / WIN) * WW + (wx * WIN + r % WIN);
    const unsigned short* p = qkv + ((long)b * HWSZ + hw) * (3 * DD);
    sq += bf2f(p[c]);
    sk += bf2f(p[DD + c]);
  }
  qr[(long)bn * DD + c] = sq * (1.f / 49.f);
  kr[(long)bn * DD + c] = sk * (1.f / 49.f);
}

__global__ void k_aff_topk(const float* __restrict__ qr,
                           const float* __restrict__ kr, int* __restrict__ idx) {
  __shared__ float aff[NWND];
  int bn = blockIdx.x;
  int b = bn >> 6;
  int m = threadIdx.x;            // 0..63
  const float* qp = qr + (long)bn * DD;
  const float* kp = kr + ((long)(b * NWND + m)) * DD;
  float s = 0.f;
  for (int c = 0; c < DD; ++c) s += qp[c] * kp[c];
  aff[m] = s;
  __syncthreads();
  if (m == 0) {
    int* op = idx + bn * TOPK;
    for (int t = 0; t < TOPK; ++t) {            // largest, ties -> lower index
      float best = -3.4e38f; int bi = 0;
      for (int j = 0; j < NWND; ++j)
        if (aff[j] > best) { best = aff[j]; bi = j; }
      op[t] = bi; aff[bi] = -3.4e38f;
    }
  }
}

// ------------------------- attention kernel -------------------------------
// One wave per (b, head, window, query-row-tile).
__global__ void __launch_bounds__(32)
k_attn(const unsigned short* __restrict__ qkv,
       const int* __restrict__ idx,
       unsigned short* __restrict__ msg) {
  __shared__ float sc[16][KPAD];             // f32 scores / exp scratch
  __shared__ unsigned short pb[16][KPAD];    // bf16 probabilities
  __shared__ unsigned short vbuf[32][DH];    // staged V chunk (32 keys x 32 dims)
  const int lane = threadIdx.x & 31;
  const int half = lane >> 4, lm = lane & 15;
  const int mtile = blockIdx.x;              // 0..3
  const int n = blockIdx.y;                  // window 0..63
  const int bh = blockIdx.z;
  const int b = bh >> 3, head = bh & 7;
  const int wy = n >> 3, wx = n & 7;
  const int* idp = idx + (b * NWND + n) * TOPK;
  const int hoff = head * DH;

  // A = q tile [16 x 32], rows >= 49 zero-padded.  Dims are contiguous (NHWC).
  v16u aq = {};
  {
    int m = mtile * 16 + lm;
    if (m < WWIN) {
      int hw = (wy * WIN + m / WIN) * WW + (wx * WIN + m % WIN);
      const unsigned short* qrow = qkv + ((long)b * HWSZ + hw) * (3 * DD) + hoff;
      aq = concat8(*(const v8u*)(qrow + half * 16),
                   *(const v8u*)(qrow + half * 16 + 8));
    }
  }

  // scores: 25 N-tiles of 16 gathered keys; k row is one contiguous 32B load.
  const float scale = 0.17677669529663687f;  // 1/sqrt(32)
  for (int nt = 0; nt < 25; ++nt) {
    v16u bf = {};
    int kk = nt * 16 + lm;
    if (kk < KLEN) {
      int t = kk / WWIN, r2 = kk % WWIN;
      int j = idp[t];
      int hw2 = ((j >> 3) * WIN + r2 / WIN) * WW + ((j & 7) * WIN + r2 % WIN);
      const unsigned short* krow = qkv + ((long)b * HWSZ + hw2) * (3 * DD) + DD + hoff;
      bf = *(const v16u*)(krow + half * 16);
    }
    v8f c = {};
    c = wmma_bf16(aq, bf, c);
#pragma unroll
    for (int i = 0; i < 8; ++i)
      sc[half * 8 + i][nt * 16 + lm] = c[i] * scale;
  }
  __syncthreads();

  // softmax over 392 real keys; write bf16 probs, zero the pad
  if (lane < 16) {
    int row = lane;
    float mx = -3.4e38f;
    for (int c = 0; c < KLEN; ++c) mx = fmaxf(mx, sc[row][c]);
    float s = 0.f;
    for (int c = 0; c < KLEN; ++c) { float e = __expf(sc[row][c] - mx); sc[row][c] = e; s += e; }
    float inv = 1.f / s;
    for (int c = 0; c < KLEN; ++c) pb[row][c] = f2bf(sc[row][c] * inv);
    for (int c = KLEN; c < KPAD; ++c) pb[row][c] = 0;
  }
  __syncthreads();

  // out = P(16x416) * V(416x32): stage each 32-key V chunk into LDS with
  // coalesced b128 loads, then transposed ds reads for the B fragment.
  v8f co0 = {}, co1 = {};
  for (int kc = 0; kc < 13; ++kc) {
    {
      int kk = kc * 32 + lane;
      v16u d0 = {}, d1 = {};
      if (kk < KLEN) {
        int t = kk / WWIN, r2 = kk % WWIN;
        int j = idp[t];
        int hw2 = ((j >> 3) * WIN + r2 / WIN) * WW + ((j & 7) * WIN + r2 % WIN);
        const unsigned short* vrow = qkv + ((long)b * HWSZ + hw2) * (3 * DD) + 2 * DD + hoff;
        d0 = *(const v16u*)(vrow);
        d1 = *(const v16u*)(vrow + 16);
      }
      *(v16u*)&vbuf[lane][0]  = d0;
      *(v16u*)&vbuf[lane][16] = d1;
    }
    __syncthreads();   // single wave: LDS pipeline is in-order per wave

    // A fragment: contiguous bf16 prob reads from LDS
    const unsigned short* prow = &pb[lm][kc * 32];
    v16u a2 = concat8(*(const v8u*)(prow + half * 16),
                      *(const v8u*)(prow + half * 16 + 8));
#pragma unroll
    for (int ndt = 0; ndt < 2; ++ndt) {
      int dd = ndt * 16 + lm;
      v16u bmf;
#pragma unroll
      for (int i = 0; i < 8; ++i) {
        int k = half * 16 + i * 2;
        bmf[2 * i]     = vbuf[k][dd];
        bmf[2 * i + 1] = vbuf[k + 1][dd];
      }
      if (ndt == 0) co0 = wmma_bf16(a2, bmf, co0);
      else          co1 = wmma_bf16(a2, bmf, co1);
    }
    __syncthreads();
  }

  // scatter to msg NHWC [b][hw][256]; lanes contiguous in dim -> coalesced
#pragma unroll
  for (int i = 0; i < 8; ++i) {
    int m = mtile * 16 + half * 8 + i;
    if (m < WWIN) {
      int hw = (wy * WIN + m / WIN) * WW + (wx * WIN + m % WIN);
      unsigned short* orow = msg + ((long)b * HWSZ + hw) * DD + hoff;
      orow[lm]      = f2bf(co0[i]);
      orow[16 + lm] = f2bf(co1[i]);
    }
  }
}

// ------------------------- merge GEMM + residual --------------------------
// x2 = x + W_merge * msg + b_merge   (NHWC f32 + NHWC bf16)
__global__ void __launch_bounds__(32)
k_gemm_merge(const unsigned short* __restrict__ Wm,
             const unsigned short* __restrict__ msg,
             const float* __restrict__ bmerge,
             const float* __restrict__ x_t,
             float* __restrict__ x2f, unsigned short* __restrict__ x2b) {
  const int lane = threadIdx.x & 31;
  const int half = lane >> 4, lm = lane & 15;
  const int n0 = blockIdx.x * 32, m0 = blockIdx.y * 64, b = blockIdx.z;
  const unsigned short* Bcol0 = msg + ((long)b * HWSZ + n0 + lm) * DD;
  const unsigned short* Bcol1 = Bcol0 + (long)16 * DD;
  v8f acc[8] = {};
  gemm_tile4x2<DD>(Wm + (long)m0 * DD, Bcol0, Bcol1, acc, lane);
#pragma unroll
  for (int nt = 0; nt < 2; ++nt) {
    long rowoff = ((long)b * HWSZ + n0 + nt * 16 + lm) * DD + m0 + half * 8;
#pragma unroll
    for (int mt = 0; mt < 4; ++mt) {
      v8f xv = *(const v8f*)(x_t + rowoff + mt * 16);
      v8f bm = *(const v8f*)(bmerge + m0 + mt * 16 + half * 8);
      v8f r;
      v8u o;
#pragma unroll
      for (int i = 0; i < 8; ++i) {
        float t = acc[mt * 2 + nt][i] + bm[i] + xv[i];
        r[i] = t;
        o[i] = f2bf(t);
      }
      *(v8f*)(x2f + rowoff + mt * 16) = r;
      *(v8u*)(x2b + rowoff + mt * 16) = o;
    }
  }
}

// ------------------------- MLP GEMM 1 (fused BN2+GELU) --------------------
__global__ void __launch_bounds__(32)
k_gemm_mlp1(const unsigned short* __restrict__ W1,
            const unsigned short* __restrict__ x2b,
            const float* __restrict__ g, const float* __restrict__ bb,
            const float* __restrict__ mm, const float* __restrict__ vv,
            unsigned short* __restrict__ hA) {
  const int lane = threadIdx.x & 31;
  const int half = lane >> 4, lm = lane & 15;
  const int n0 = blockIdx.x * 32, m0 = blockIdx.y * 64, b = blockIdx.z;
  const unsigned short* Bcol0 = x2b + ((long)b * HWSZ + n0 + lm) * DD;
  const unsigned short* Bcol1 = Bcol0 + (long)16 * DD;
  v8f acc[8] = {};
  gemm_tile4x2<DD>(W1 + (long)m0 * DD, Bcol0, Bcol1, acc, lane);
#pragma unroll
  for (int nt = 0; nt < 2; ++nt) {
    unsigned short* orow =
        hA + ((long)b * HWSZ + n0 + nt * 16 + lm) * DM + m0 + half * 8;
#pragma unroll
    for (int mt = 0; mt < 4; ++mt) {
      int mb = m0 + mt * 16 + half * 8;
      v8f gv = *(const v8f*)(g + mb),  bv = *(const v8f*)(bb + mb);
      v8f mv = *(const v8f*)(mm + mb), vvv = *(const v8f*)(vv + mb);
      v8u o;
#pragma unroll
      for (int i = 0; i < 8; ++i)
        o[i] = f2bf(gelu_exact(bnv(acc[mt * 2 + nt][i], gv[i], bv[i], mv[i], vvv[i])));
      *(v8u*)(orow + mt * 16) = o;
    }
  }
}

// ------------------------- depthwise 3x3 + BN3 + GELU (NHWC) --------------
__global__ void k_dwconv(const unsigned short* __restrict__ hA,
                         const float* __restrict__ Wdw,
                         const float* __restrict__ g, const float* __restrict__ bb,
                         const float* __restrict__ mm, const float* __restrict__ vv,
                         unsigned short* __restrict__ h3) {
  long i = (long)blockIdx.x * 256 + threadIdx.x;          // [b][hw][c], c fastest
  if (i >= (long)BB * HWSZ * DM) return;
  int c = (int)(i & (DM - 1));
  long t = i >> 10;
  int hw = (int)(t % HWSZ);
  int b = (int)(t / HWSZ);
  int y = hw / WW, x = hw % WW;
  float s = 0.f;
#pragma unroll
  for (int ky = 0; ky < 3; ++ky)
#pragma unroll
    for (int kx = 0; kx < 3; ++kx) {
      int yy = y + ky - 1, xx = x + kx - 1;
      if (yy >= 0 && yy < HH && xx >= 0 && xx < WW)
        s += bf2f(hA[((long)b * HWSZ + yy * WW + xx) * DM + c]) * Wdw[c * 9 + ky * 3 + kx];
    }
  h3[i] = f2bf(gelu_exact(bnv(s, g[c], bb[c], mm[c], vv[c])));
}

// ------------------------- MLP GEMM 2 (fused BN4 + residual) --------------
// out (NCHW f32, final) = x2 + bn4(W2 * h3)
__global__ void __launch_bounds__(32)
k_gemm_mlp2(const unsigned short* __restrict__ W2,
            const unsigned short* __restrict__ h3,
            const float* __restrict__ g, const float* __restrict__ bb,
            const float* __restrict__ mm, const float* __restrict__ vv,
            const float* __restrict__ x2f,
            float* __restrict__ out) {
  const int lane = threadIdx.x & 31;
  const int half = lane >> 4, lm = lane & 15;
  const int n0 = blockIdx.x * 32, m0 = blockIdx.y * 64, b = blockIdx.z;
  const unsigned short* Bcol0 = h3 + ((long)b * HWSZ + n0 + lm) * DM;
  const unsigned short* Bcol1 = Bcol0 + (long)16 * DM;
  v8f acc[8] = {};
  gemm_tile4x2<DM>(W2 + (long)m0 * DM, Bcol0, Bcol1, acc, lane);
#pragma unroll
  for (int nt = 0; nt < 2; ++nt) {
    const int n = n0 + nt * 16 + lm;
#pragma unroll
    for (int mt = 0; mt < 4; ++mt) {
      int mb = m0 + mt * 16 + half * 8;
      v8f xv = *(const v8f*)(x2f + ((long)b * HWSZ + n) * DD + mb);
      v8f gv = *(const v8f*)(g + mb),  bv = *(const v8f*)(bb + mb);
      v8f mv = *(const v8f*)(mm + mb), vvv = *(const v8f*)(vv + mb);
#pragma unroll
      for (int i = 0; i < 8; ++i)
        out[((long)b * DD + mb + i) * HWSZ + n] =
            xv[i] + bnv(acc[mt * 2 + nt][i], gv[i], bv[i], mv[i], vvv[i]);
    }
  }
}

// ---------------------------------------------------------------------------
extern "C" void kernel_launch(void* const* d_in, const int* in_sizes, int n_in,
                              void* d_out, int out_size, void* d_ws, size_t ws_size,
                              hipStream_t stream) {
  const float* x      = (const float*)d_in[0];
  const float* g1     = (const float*)d_in[1];
  const float* b1     = (const float*)d_in[2];
  const float* m1     = (const float*)d_in[3];
  const float* v1     = (const float*)d_in[4];
  const float* W_qkv  = (const float*)d_in[5];
  const float* W_mrg  = (const float*)d_in[6];
  const float* b_mrg  = (const float*)d_in[7];
  const float* W1     = (const float*)d_in[8];
  const float* g2     = (const float*)d_in[9];
  const float* b2     = (const float*)d_in[10];
  const float* m2     = (const float*)d_in[11];
  const float* v2     = (const float*)d_in[12];
  const float* Wdw    = (const float*)d_in[13];
  const float* g3     = (const float*)d_in[14];
  const float* b3     = (const float*)d_in[15];
  const float* m3     = (const float*)d_in[16];
  const float* v3     = (const float*)d_in[17];
  const float* W2     = (const float*)d_in[18];
  const float* g4     = (const float*)d_in[19];
  const float* b4     = (const float*)d_in[20];
  const float* m4     = (const float*)d_in[21];
  const float* v4     = (const float*)d_in[22];
  float* out = (float*)d_out;

  // ---- workspace bump allocator (256B aligned) ----
  char* wp = (char*)d_ws;
  auto alloc = [&](size_t bytes) -> void* {
    void* p = wp;
    wp += (bytes + 255) & ~(size_t)255;
    return p;
  };
  float*          x_t    = (float*)alloc((size_t)BB * HWSZ * DD * 4);          // NHWC raw x
  unsigned short* h_bf   = (unsigned short*)alloc((size_t)BB * HWSZ * DD * 2); // NHWC
  unsigned short* qkv_bf = (unsigned short*)alloc((size_t)BB * HWSZ * 3 * DD * 2);
  unsigned short* Wq_bf  = (unsigned short*)alloc((size_t)3 * DD * DD * 2);
  unsigned short* Wm_bf  = (unsigned short*)alloc((size_t)DD * DD * 2);
  unsigned short* W1_bf  = (unsigned short*)alloc((size_t)DM * DD * 2);
  unsigned short* W2_bf  = (unsigned short*)alloc((size_t)DD * DM * 2);
  float*          qr     = (float*)alloc((size_t)BB * NWND * DD * 4);
  float*          kr     = (float*)alloc((size_t)BB * NWND * DD * 4);
  int*            idx    = (int*)alloc((size_t)BB * NWND * TOPK * 4);
  unsigned short* msg_bf = (unsigned short*)alloc((size_t)BB * HWSZ * DD * 2);
  float*          x2f    = (float*)alloc((size_t)BB * HWSZ * DD * 4);
  unsigned short* x2b    = (unsigned short*)alloc((size_t)BB * HWSZ * DD * 2);
  unsigned short* hA_bf  = (unsigned short*)alloc((size_t)BB * HWSZ * DM * 2);
  unsigned short* h3_bf  = (unsigned short*)alloc((size_t)BB * HWSZ * DM * 2);

  const int NT2 = HWSZ / 32;   // 98 column tiles of 32

  // weight conversions to bf16
  k_f32_to_bf16<<<(3 * DD * DD + 255) / 256, 256, 0, stream>>>(W_qkv, Wq_bf, 3 * DD * DD);
  k_f32_to_bf16<<<(DD * DD + 255) / 256, 256, 0, stream>>>(W_mrg, Wm_bf, DD * DD);
  k_f32_to_bf16<<<(DM * DD + 255) / 256, 256, 0, stream>>>(W1, W1_bf, DM * DD);
  k_f32_to_bf16<<<(DD * DM + 255) / 256, 256, 0, stream>>>(W2, W2_bf, DD * DM);

  // BN1 + GELU -> NHWC bf16 (+ raw NHWC f32 for the residual)
  {
    long n = (long)BB * HWSZ * DD;
    k_bn_gelu_x<<<(unsigned)((n + 255) / 256), 256, 0, stream>>>(x, g1, b1, m1, v1, x_t, h_bf);
  }

  // qkv projection GEMM
  k_gemm_qkv<<<dim3(NT2, 3 * DD / 64, BB), 32, 0, stream>>>(Wq_bf, h_bf, qkv_bf);

  // routing
  k_pool<<<BB * NWND, DD, 0, stream>>>(qkv_bf, qr, kr);
  k_aff_topk<<<BB * NWND, NWND, 0, stream>>>(qr, kr, idx);

  // windowed top-k attention
  k_attn<<<dim3(4, NWND, BB * HEADS), 32, 0, stream>>>(qkv_bf, idx, msg_bf);

  // merge + residual
  k_gemm_merge<<<dim3(NT2, DD / 64, BB), 32, 0, stream>>>(Wm_bf, msg_bf, b_mrg, x_t, x2f, x2b);

  // MLP
  k_gemm_mlp1<<<dim3(NT2, DM / 64, BB), 32, 0, stream>>>(W1_bf, x2b, g2, b2, m2, v2, hA_bf);
  {
    long n = (long)BB * HWSZ * DM;
    k_dwconv<<<(unsigned)((n + 255) / 256), 256, 0, stream>>>(hA_bf, Wdw, g3, b3, m3, v3, h3_bf);
  }
  k_gemm_mlp2<<<dim3(NT2, DD / 64, BB), 32, 0, stream>>>(W2_bf, h3_bf, g4, b4, m4, v4, x2f, out);

  (void)in_sizes; (void)n_in; (void)out_size; (void)ws_size;
}
